// UniversalReasoner_50251117363362
// MI455X (gfx1250) — matrix operand order repaired and checked by reading, not compiled, not run on hardware
//
#include <hip/hip_runtime.h>
#include <hip/hip_bf16.h>
#include <math.h>

// ---------------------------------------------------------------------------
// UniversalReasoner for MI455X (gfx1250, wave32, WMMA).
// Dense matmuls: V_WMMA_F32_16X16X32_BF16, bf16 operands / f32 accumulate.
// Weights pre-swizzled into WMMA B-fragment layout (2x b128 loads per frag).
// Activations produced directly in bf16 row-major (2x b128 A-frag loads).
// ---------------------------------------------------------------------------

#define D_MODEL   1024
#define N_HEADS   16
#define N_GROUPS  4
#define HEAD_DIM  64
#define SEQ_LEN   512
#define N_SLOTS   32
#define FFN_DIM   2816
#define VOCAB_SZ  100352
#define N_STEPS   4
#define ROPE_ROWS (SEQ_LEN + N_SLOTS)

typedef __attribute__((ext_vector_type(16))) __bf16 v16bf;
typedef __attribute__((ext_vector_type(8)))  float  v8f;

union BF16Frag {
  v16bf v;
  __bf16 b[16];
  unsigned short h[16];
  uint4 q[2];
};

__device__ __forceinline__ unsigned short f2bfu(float f) {
  __bf16 b = (__bf16)f;
  return __builtin_bit_cast(unsigned short, b);
}
__device__ __forceinline__ float bf2f(unsigned short u) {
  return __uint_as_float((unsigned int)u << 16);
}

// Epilogue: mode 0: C = AB + bias; 1: C += AB + bias; 2: C = sigmoid(AB + bias)
template<int NT>
__device__ __forceinline__ void store_tile(float* __restrict__ C,
                                           const float* __restrict__ bias,
                                           v8f (&acc)[NT], int m0, int n0,
                                           int r, int half, int N, int mode) {
#pragma unroll
  for (int t = 0; t < NT; ++t) {
    const int col = n0 + t * 16 + r;
    const float b = bias ? bias[col] : 0.0f;
#pragma unroll
    for (int vv = 0; vv < 8; ++vv) {
      const int row = m0 + vv + half * 8;
      const size_t idx = (size_t)row * N + col;
      float out = acc[t][vv] + b;
      if (mode == 1)      out += C[idx];
      else if (mode == 2) out = 1.0f / (1.0f + __expf(-out));
      C[idx] = out;
    }
  }
}

// A fragment (bf16 row-major [M][K]) : lane r=lane&15 -> row m0+r,
// K = {kb..kb+15}, kb=(lane>>4)*16  => two 16B loads.
__device__ __forceinline__ void load_afrag(const unsigned short* __restrict__ arow,
                                           int koff, BF16Frag& af) {
  af.q[0] = *(const uint4*)(arow + koff);
  af.q[1] = *(const uint4*)(arow + koff + 8);
}

// ---------------------------------------------------------------------------
// GEMM: A bf16 [M,K] x pre-swizzled weight fragments -> C f32 [M,N]
// WB fragment (nt,kt): 32 lanes x 16 bf16 at ((nt*(K/32)+kt)*32+lane)*16
// grid.x = M/16 (fast), grid.y = N/(16*NT) (L2 reuse of W slices across M).
// ---------------------------------------------------------------------------
template<int NT>
__global__ __launch_bounds__(32)
void k_gemm_aws(const unsigned short* __restrict__ A,
                const unsigned short* __restrict__ WB,
                const float* __restrict__ bias, float* __restrict__ C,
                int M, int N, int K, int mode) {
  const int lane = threadIdx.x;
  const int m0 = blockIdx.x * 16;
  const int ng = blockIdx.y;
  const int r = lane & 15, half = lane >> 4;
  const int KT = K >> 5;
  const unsigned short* arow = A + (size_t)(m0 + r) * K + half * 16;
  v8f acc[NT] = {};
  for (int kt = 0; kt < KT; ++kt) {
    BF16Frag af, bf[NT];
    load_afrag(arow, kt * 32, af);
#pragma unroll
    for (int t = 0; t < NT; ++t) {
      const uint4* bp = (const uint4*)(WB +
          (((size_t)(ng * NT + t) * KT + kt) * 32 + lane) * 16);
      bf[t].q[0] = bp[0];
      bf[t].q[1] = bp[1];
    }
    __builtin_prefetch(arow + kt * 32 + 32, 0, 0);
#pragma unroll
    for (int t = 0; t < NT; ++t)
      acc[t] = __builtin_amdgcn_wmma_f32_16x16x32_bf16(
          false, af.v, false, bf[t].v, (short)0, acc[t], false, false);
  }
  store_tile<NT>(C, bias, acc, m0, ng * (16 * NT), r, half, N, mode);
}

// Fallback: A bf16 x W f32 [K,N] strided (batched loads, cvt in regs).
template<int NT>
__global__ __launch_bounds__(32)
void k_gemm_anf(const unsigned short* __restrict__ A, const float* __restrict__ W,
                const float* __restrict__ bias, float* __restrict__ C,
                int M, int N, int K, int mode) {
  const int lane = threadIdx.x;
  const int m0 = blockIdx.x * 16;
  const int ng = blockIdx.y;
  const int r = lane & 15, half = lane >> 4;
  const unsigned short* arow = A + (size_t)(m0 + r) * K + half * 16;
  v8f acc[NT] = {};
  for (int k0 = 0; k0 < K; k0 += 32) {
    BF16Frag af, bf[NT];
    load_afrag(arow, k0, af);
#pragma unroll
    for (int t = 0; t < NT; ++t) {
      const float* wp = W + (size_t)(k0 + half * 16) * N + ng * (16 * NT) + t * 16 + r;
      float f[16];
#pragma unroll
      for (int i = 0; i < 16; ++i) f[i] = wp[(size_t)i * N];
#pragma unroll
      for (int i = 0; i < 16; ++i) bf[t].b[i] = (__bf16)f[i];
    }
#pragma unroll
    for (int t = 0; t < NT; ++t)
      acc[t] = __builtin_amdgcn_wmma_f32_16x16x32_bf16(
          false, af.v, false, bf[t].v, (short)0, acc[t], false, false);
  }
  store_tile<NT>(C, bias, acc, m0, ng * (16 * NT), r, half, N, mode);
}

// Head: A bf16 x Bt f32 [N,K] row-major (embed; K-contiguous, batched float4).
template<int NT>
__global__ __launch_bounds__(32)
void k_gemm_abt_f32(const unsigned short* __restrict__ A, const float* __restrict__ Bt,
                    float* __restrict__ C, int M, int N, int K) {
  const int lane = threadIdx.x;
  const int m0 = blockIdx.x * 16;
  const int ng = blockIdx.y;
  const int r = lane & 15, half = lane >> 4;
  const unsigned short* arow = A + (size_t)(m0 + r) * K + half * 16;
  v8f acc[NT] = {};
  for (int k0 = 0; k0 < K; k0 += 32) {
    BF16Frag af, bf[NT];
    load_afrag(arow, k0, af);
#pragma unroll
    for (int t = 0; t < NT; ++t) {
      const float4* bp = (const float4*)(Bt +
          (size_t)(ng * (16 * NT) + t * 16 + r) * K + k0 + half * 16);
      float4 f0 = bp[0], f1 = bp[1], f2 = bp[2], f3 = bp[3];
      bf[t].b[0]  = (__bf16)f0.x; bf[t].b[1]  = (__bf16)f0.y;
      bf[t].b[2]  = (__bf16)f0.z; bf[t].b[3]  = (__bf16)f0.w;
      bf[t].b[4]  = (__bf16)f1.x; bf[t].b[5]  = (__bf16)f1.y;
      bf[t].b[6]  = (__bf16)f1.z; bf[t].b[7]  = (__bf16)f1.w;
      bf[t].b[8]  = (__bf16)f2.x; bf[t].b[9]  = (__bf16)f2.y;
      bf[t].b[10] = (__bf16)f2.z; bf[t].b[11] = (__bf16)f2.w;
      bf[t].b[12] = (__bf16)f3.x; bf[t].b[13] = (__bf16)f3.y;
      bf[t].b[14] = (__bf16)f3.z; bf[t].b[15] = (__bf16)f3.w;
    }
#pragma unroll
    for (int t = 0; t < NT; ++t)
      acc[t] = __builtin_amdgcn_wmma_f32_16x16x32_bf16(
          false, af.v, false, bf[t].v, (short)0, acc[t], false, false);
  }
  store_tile<NT>(C, (const float*)nullptr, acc, m0, ng * (16 * NT), r, half, N, 0);
}

// A bf16 x Bt bf16 [N,K] row-major (InfoNCE a @ a^T). Pure b128 loads.
template<int NT>
__global__ __launch_bounds__(32)
void k_gemm_abt_bf(const unsigned short* __restrict__ A, const unsigned short* __restrict__ Bt,
                   float* __restrict__ C, int M, int N, int K) {
  const int lane = threadIdx.x;
  const int m0 = blockIdx.x * 16;
  const int ng = blockIdx.y;
  const int r = lane & 15, half = lane >> 4;
  const unsigned short* arow = A + (size_t)(m0 + r) * K + half * 16;
  v8f acc[NT] = {};
  for (int k0 = 0; k0 < K; k0 += 32) {
    BF16Frag af, bf[NT];
    load_afrag(arow, k0, af);
#pragma unroll
    for (int t = 0; t < NT; ++t) {
      const uint4* bp = (const uint4*)(Bt +
          (size_t)(ng * (16 * NT) + t * 16 + r) * K + k0 + half * 16);
      bf[t].q[0] = bp[0];
      bf[t].q[1] = bp[1];
    }
#pragma unroll
    for (int t = 0; t < NT; ++t)
      acc[t] = __builtin_amdgcn_wmma_f32_16x16x32_bf16(
          false, af.v, false, bf[t].v, (short)0, acc[t], false, false);
  }
  store_tile<NT>(C, (const float*)nullptr, acc, m0, ng * (16 * NT), r, half, N, 0);
}

// Swizzle a weight W[K][N] (f32) into bf16 B-fragments.
// One thread per (nt,kt,lane): writes 16 bf16 (32B contiguous).
__global__ void k_swizzle_w(const float* __restrict__ W, unsigned short* __restrict__ WB,
                            int K, int N) {
  const int tid = blockIdx.x * 256 + threadIdx.x;
  const int KT = K >> 5;
  const int total = (N >> 4) * KT * 32;
  if (tid >= total) return;
  const int lane = tid & 31;
  const int frag = tid >> 5;
  const int kt = frag % KT;
  const int nt = frag / KT;
  const int col = nt * 16 + (lane & 15);
  const int kbase = kt * 32 + (lane >> 4) * 16;
  float f[16];
#pragma unroll
  for (int i = 0; i < 16; ++i) f[i] = W[(size_t)(kbase + i) * N + col];
  BF16Frag o;
#pragma unroll
  for (int i = 0; i < 16; ++i) o.b[i] = (__bf16)f[i];
  uint4* dst = (uint4*)(WB + (size_t)tid * 16);
  dst[0] = o.q[0];
  dst[1] = o.q[1];
}

// ---------------------------------------------------------------------------
// Small kernels
// ---------------------------------------------------------------------------
__global__ void k_gather(const int* __restrict__ tokens,
                         const float* __restrict__ embed, float* __restrict__ x) {
  const int s = blockIdx.x;
  const int t = tokens[s];
  for (int d = threadIdx.x; d < D_MODEL; d += 256)
    x[(size_t)s * D_MODEL + d] = embed[(size_t)t * D_MODEL + d];
}

__global__ void k_rope_tables(float* __restrict__ cosT, float* __restrict__ sinT) {
  const int t = blockIdx.x;
  const int i = threadIdx.x; // 0..31
  const float inv = __powf(10000.0f, -(2.0f * i) / (float)HEAD_DIM);
  const float f = (float)t * inv;
  cosT[t * 32 + i] = __cosf(f);
  sinT[t * 32 + i] = __sinf(f);
}

// y_f32 = x * rsqrt((mean?sum/dim:sum)+eps) * (scale or 1)
__global__ void k_rmsnorm(const float* __restrict__ x, const float* __restrict__ scale,
                          float* __restrict__ y, int dim, float eps, int mean_flag) {
  const int row = blockIdx.x, tid = threadIdx.x;
  const float* xr = x + (size_t)row * dim;
  float* yr = y + (size_t)row * dim;
  float ss = 0.0f;
  for (int i = tid; i < dim; i += 256) { float v = xr[i]; ss += v * v; }
  __shared__ float red[256];
  red[tid] = ss; __syncthreads();
  for (int s = 128; s > 0; s >>= 1) { if (tid < s) red[tid] += red[tid + s]; __syncthreads(); }
  const float denom = mean_flag ? (red[0] / (float)dim) : red[0];
  const float inv = rsqrtf(denom + eps);
  for (int i = tid; i < dim; i += 256) {
    const float sv = scale ? scale[i] : 1.0f;
    yr[i] = xr[i] * inv * sv;
  }
}

// bf16-output variant (feeds WMMA A operands).
__global__ void k_rmsnorm_bf(const float* __restrict__ x, const float* __restrict__ scale,
                             unsigned short* __restrict__ y, int dim, float eps, int mean_flag) {
  const int row = blockIdx.x, tid = threadIdx.x;
  const float* xr = x + (size_t)row * dim;
  unsigned short* yr = y + (size_t)row * dim;
  float ss = 0.0f;
  for (int i = tid; i < dim; i += 256) { float v = xr[i]; ss += v * v; }
  __shared__ float red[256];
  red[tid] = ss; __syncthreads();
  for (int s = 128; s > 0; s >>= 1) { if (tid < s) red[tid] += red[tid + s]; __syncthreads(); }
  const float denom = mean_flag ? (red[0] / (float)dim) : red[0];
  const float inv = rsqrtf(denom + eps);
  for (int i = tid; i < dim; i += 256) {
    const float sv = scale ? scale[i] : 1.0f;
    yr[i] = f2bfu(xr[i] * inv * sv);
  }
}

// Per-head RMS (qn/kn), RoPE (pos = row index), optional 1/sqrt(HD) scale.
__global__ void k_qkv_post(float* __restrict__ buf, int nheads,
                           const float* __restrict__ hs,
                           const float* __restrict__ cosT,
                           const float* __restrict__ sinT, float postscale) {
  const int row = blockIdx.x, h = blockIdx.y, lane = threadIdx.x; // wave32
  float* p = buf + (size_t)row * (nheads * HEAD_DIM) + h * HEAD_DIM;
  float x1 = p[lane], x2 = p[lane + 32];
  float ss = x1 * x1 + x2 * x2;
  for (int m = 16; m > 0; m >>= 1) ss += __shfl_xor(ss, m, 32);
  const float inv = rsqrtf(ss / 64.0f + 1e-6f);
  x1 *= inv * hs[lane];
  x2 *= inv * hs[lane + 32];
  const float c = cosT[row * 32 + lane], s = sinT[row * 32 + lane];
  p[lane]      = (x1 * c - x2 * s) * postscale;
  p[lane + 32] = (x1 * s + x2 * c) * postscale;
}

// Softmax attention with GQA. One block (128 thr) per (query row, head).
// Output o written bf16 (it is the A operand of the o@ow GEMM).
__global__ void k_attn(const float* __restrict__ q, const float* __restrict__ k,
                       const float* __restrict__ v, unsigned short* __restrict__ o,
                       int Sk, int causal) {
  const int qi = blockIdx.x, h = blockIdx.y, tid = threadIdx.x;
  const int g = h >> 2; // NH/NG = 4
  __shared__ float lp[SEQ_LEN];
  __shared__ float red[128];
  const float* qv = q + (size_t)qi * D_MODEL + h * HEAD_DIM;
  for (int kk = tid; kk < Sk; kk += 128) {
    const float* kv = k + (size_t)kk * (N_GROUPS * HEAD_DIM) + g * HEAD_DIM;
    float acc = 0.0f;
#pragma unroll 8
    for (int d = 0; d < HEAD_DIM; ++d) acc += qv[d] * kv[d];
    if (causal && kk > qi) acc = -1e9f; // keep when q_pos >= kv_pos
    lp[kk] = acc;
  }
  __syncthreads();
  float lm = -3.0e38f;
  for (int kk = tid; kk < Sk; kk += 128) lm = fmaxf(lm, lp[kk]);
  red[tid] = lm; __syncthreads();
  for (int s = 64; s > 0; s >>= 1) { if (tid < s) red[tid] = fmaxf(red[tid], red[tid + s]); __syncthreads(); }
  const float m = red[0]; __syncthreads();
  float ls = 0.0f;
  for (int kk = tid; kk < Sk; kk += 128) { float e = __expf(lp[kk] - m); lp[kk] = e; ls += e; }
  red[tid] = ls; __syncthreads();
  for (int s = 64; s > 0; s >>= 1) { if (tid < s) red[tid] += red[tid + s]; __syncthreads(); }
  const float sinv = 1.0f / red[0];
  __syncthreads();
  if (tid < HEAD_DIM) {
    float acc = 0.0f;
    for (int kk = 0; kk < Sk; ++kk)
      acc += lp[kk] * v[(size_t)kk * (N_GROUPS * HEAD_DIM) + g * HEAD_DIM + tid];
    o[(size_t)qi * D_MODEL + h * HEAD_DIM + tid] = f2bfu(acc * sinv);
  }
}

__global__ void k_silu_mul_bf(const float* __restrict__ g, const float* __restrict__ u,
                              unsigned short* __restrict__ out, int n) {
  const int i = blockIdx.x * 256 + threadIdx.x;
  if (i < n) {
    const float x = g[i];
    out[i] = f2bfu((x / (1.0f + __expf(-x))) * u[i]);
  }
}

__global__ void k_copy(float* __restrict__ dst, const float* __restrict__ src, int n) {
  const int i = blockIdx.x * 256 + threadIdx.x;
  if (i < n) dst[i] = src[i];
}

__global__ void k_add_rowvec(float* __restrict__ dst, const float* __restrict__ src,
                             const float* __restrict__ vec, int n) {
  const int i = blockIdx.x * 256 + threadIdx.x;
  if (i < n) dst[i] = src[i] + vec[i & (D_MODEL - 1)];
}

__global__ void k_blend(float* __restrict__ ns, const float* __restrict__ sh,
                        const float* __restrict__ h, const float* __restrict__ gate, int n) {
  const int i = blockIdx.x * 256 + threadIdx.x;
  if (i < n) { const float g = gate[i]; ns[i] = sh[i] * (1.0f - g) + h[i] * g; }
}

__global__ void k_zero(float* __restrict__ p, int n) {
  const int i = blockIdx.x * 256 + threadIdx.x;
  if (i < n) p[i] = 0.0f;
}

// InfoNCE step loss accumulated into *div. One wave, one lane per slot.
__global__ void k_infonce(const unsigned short* __restrict__ a,
                          const unsigned short* __restrict__ pn,
                          const float* __restrict__ neg,
                          const float* __restrict__ raw_tau, float* __restrict__ div) {
  const int i = threadIdx.x; // 0..31
  const float tau = __expf(*raw_tau);
  float pos = 0.0f;
  for (int d = 0; d < D_MODEL; ++d)
    pos += bf2f(a[i * D_MODEL + d]) * bf2f(pn[i * D_MODEL + d]);
  pos /= tau;
  float m = pos;
  for (int j = 0; j < N_SLOTS; ++j) if (j != i) m = fmaxf(m, neg[i * N_SLOTS + j] / tau);
  float sum = __expf(pos - m);
  for (int j = 0; j < N_SLOTS; ++j) if (j != i) sum += __expf(neg[i * N_SLOTS + j] / tau - m);
  float loss = -(pos - m - __logf(sum));
  for (int mm = 16; mm > 0; mm >>= 1) loss += __shfl_xor(loss, mm, 32);
  if (i == 0) *div += loss / (float)N_SLOTS;
}

__global__ void k_finalize(const float* __restrict__ div, float* __restrict__ out) {
  out[0] = div[0] / (float)N_STEPS;
}

// ---------------------------------------------------------------------------
// Host orchestration
// ---------------------------------------------------------------------------
struct BlockP {
  const float *n1, *n2, *qn, *kn, *qw, *qb, *kw, *kb, *vw, *vb,
              *ow, *ob, *gw, *gb, *uw, *ub, *dw, *db;
};
struct WBlk { unsigned short *qw, *kw, *vw, *ow, *gw, *uw, *dw; };

static BlockP load_block(void* const* d_in, int base, bool sorted_keys) {
  const float* a[18];
  for (int i = 0; i < 18; ++i) a[i] = (const float*)d_in[base + i];
  BlockP p;
  if (sorted_keys) { // db,dw,gb,gw,kb,kn,kw,n1,n2,ob,ow,qb,qn,qw,ub,uw,vb,vw
    p.db = a[0];  p.dw = a[1];  p.gb = a[2];  p.gw = a[3];  p.kb = a[4];
    p.kn = a[5];  p.kw = a[6];  p.n1 = a[7];  p.n2 = a[8];  p.ob = a[9];
    p.ow = a[10]; p.qb = a[11]; p.qn = a[12]; p.qw = a[13]; p.ub = a[14];
    p.uw = a[15]; p.vb = a[16]; p.vw = a[17];
  } else {         // insertion: n1,n2,qn,kn,qw,qb,kw,kb,vw,vb,ow,ob,gw,gb,uw,ub,dw,db
    p.n1 = a[0];  p.n2 = a[1];  p.qn = a[2];  p.kn = a[3];  p.qw = a[4];
    p.qb = a[5];  p.kw = a[6];  p.kb = a[7];  p.vw = a[8];  p.vb = a[9];
    p.ow = a[10]; p.ob = a[11]; p.gw = a[12]; p.gb = a[13]; p.uw = a[14];
    p.ub = a[15]; p.dw = a[16]; p.db = a[17];
  }
  return p;
}

// Weight GEMM: swizzled path if Wswz != null, else strided-f32 fallback.
static void gemm_w(hipStream_t st, const unsigned short* Abf, const float* Wf32,
                   const unsigned short* Wswz, const float* bias, float* C,
                   int M, int N, int K, int mode) {
  if (Wswz) {
    if (N % 64 == 0)      { dim3 g(M/16, N/64); k_gemm_aws<4><<<g,32,0,st>>>(Abf,Wswz,bias,C,M,N,K,mode); }
    else if (N % 32 == 0) { dim3 g(M/16, N/32); k_gemm_aws<2><<<g,32,0,st>>>(Abf,Wswz,bias,C,M,N,K,mode); }
    else                  { dim3 g(M/16, N/16); k_gemm_aws<1><<<g,32,0,st>>>(Abf,Wswz,bias,C,M,N,K,mode); }
  } else {
    if (N % 64 == 0)      { dim3 g(M/16, N/64); k_gemm_anf<4><<<g,32,0,st>>>(Abf,Wf32,bias,C,M,N,K,mode); }
    else if (N % 32 == 0) { dim3 g(M/16, N/32); k_gemm_anf<2><<<g,32,0,st>>>(Abf,Wf32,bias,C,M,N,K,mode); }
    else                  { dim3 g(M/16, N/16); k_gemm_anf<1><<<g,32,0,st>>>(Abf,Wf32,bias,C,M,N,K,mode); }
  }
}

static void swz(hipStream_t st, const float* W, unsigned short* WB, int K, int N) {
  const int total = (N / 16) * (K / 32) * 32;
  k_swizzle_w<<<(total + 255) / 256, 256, 0, st>>>(W, WB, K, N);
}

struct Ws {
  float *cosT, *sinT, *x, *q, *k, *v, *g, *u;
  float *sh, *hs, *newsh, *gate, *tvec, *neg, *div;
  unsigned short *xn, *kvn, *o, *hbf, *an, *pn;
};

static void run_block(hipStream_t st, const Ws& w, const BlockP& p, const WBlk* wb,
                      float* x, int rq, const float* ctx, int rkv_, bool causal) {
  const int rkv = ctx ? rkv_ : rq;
  k_rmsnorm_bf<<<rq, 256, 0, st>>>(x, p.n1, w.xn, D_MODEL, 1e-6f, 1);
  const unsigned short* kvn = w.xn;
  if (ctx) {
    k_rmsnorm_bf<<<rkv, 256, 0, st>>>(ctx, p.n1, w.kvn, D_MODEL, 1e-6f, 1);
    kvn = w.kvn;
  }
  gemm_w(st, w.xn, p.qw, wb ? wb->qw : nullptr, p.qb, w.q, rq, D_MODEL, D_MODEL, 0);
  gemm_w(st, kvn, p.kw, wb ? wb->kw : nullptr, p.kb, w.k, rkv, N_GROUPS * HEAD_DIM, D_MODEL, 0);
  gemm_w(st, kvn, p.vw, wb ? wb->vw : nullptr, p.vb, w.v, rkv, N_GROUPS * HEAD_DIM, D_MODEL, 0);
  { dim3 g(rq, N_HEADS);   k_qkv_post<<<g, 32, 0, st>>>(w.q, N_HEADS,  p.qn, w.cosT, w.sinT, 0.125f); }
  { dim3 g(rkv, N_GROUPS); k_qkv_post<<<g, 32, 0, st>>>(w.k, N_GROUPS, p.kn, w.cosT, w.sinT, 1.0f); }
  { dim3 g(rq, N_HEADS);   k_attn<<<g, 128, 0, st>>>(w.q, w.k, w.v, w.o, rkv, causal ? 1 : 0); }
  gemm_w(st, w.o, p.ow, wb ? wb->ow : nullptr, p.ob, x, rq, D_MODEL, D_MODEL, 1);   // x += o@ow+ob
  k_rmsnorm_bf<<<rq, 256, 0, st>>>(x, p.n2, w.xn, D_MODEL, 1e-6f, 1);
  gemm_w(st, w.xn, p.gw, wb ? wb->gw : nullptr, p.gb, w.g, rq, FFN_DIM, D_MODEL, 0);
  gemm_w(st, w.xn, p.uw, wb ? wb->uw : nullptr, p.ub, w.u, rq, FFN_DIM, D_MODEL, 0);
  { const int n = rq * FFN_DIM; k_silu_mul_bf<<<(n + 255) / 256, 256, 0, st>>>(w.g, w.u, w.hbf, n); }
  gemm_w(st, w.hbf, p.dw, wb ? wb->dw : nullptr, p.db, x, rq, D_MODEL, FFN_DIM, 1); // x += h@dw+db
}

extern "C" void kernel_launch(void* const* d_in, const int* in_sizes, int n_in,
                              void* d_out, int out_size, void* d_ws, size_t ws_size,
                              hipStream_t stream) {
  (void)n_in; (void)out_size;
  const bool sorted_keys = (in_sizes[5] == FFN_DIM * D_MODEL); // d_in[5]: dw(sorted) vs n2(insertion)

  const int*   tokens   = (const int*)d_in[0];
  const float* embed    = (const float*)d_in[1];
  const float* shared0  = (const float*)d_in[2];
  const float* time_emb = (const float*)d_in[3];
  const BlockP enc = load_block(d_in, 4, sorted_keys);
  const BlockP dec = load_block(d_in, 22, sorted_keys);
  BlockP rsn[4];
  for (int i = 0; i < 4; ++i) rsn[i] = load_block(d_in, 40 + 18 * i, sorted_keys);
  const float* seq_norm   = (const float*)d_in[112];
  const float* time_norm  = (const float*)d_in[113];
  const float* hunch_norm = (const float*)d_in[114];
  const float* hgw        = (const float*)d_in[115];
  const float* hgb        = (const float*)d_in[116];
  const float* raw_tau    = (const float*)d_in[117];

  // Byte bump allocator over d_ws (256B-aligned buffers).
  char* wsb = (char*)d_ws;
  size_t off = 0;
  auto alloc = [&](size_t bytes) -> void* {
    void* p = wsb + off;
    off += (bytes + 255) & ~(size_t)255;
    return p;
  };
  Ws w;
  w.cosT = (float*)alloc((size_t)ROPE_ROWS * 32 * 4);
  w.sinT = (float*)alloc((size_t)ROPE_ROWS * 32 * 4);
  w.x    = (float*)alloc((size_t)SEQ_LEN * D_MODEL * 4);
  w.q    = (float*)alloc((size_t)SEQ_LEN * D_MODEL * 4);
  w.k    = (float*)alloc((size_t)SEQ_LEN * N_GROUPS * HEAD_DIM * 4);
  w.v    = (float*)alloc((size_t)SEQ_LEN * N_GROUPS * HEAD_DIM * 4);
  w.g    = (float*)alloc((size_t)SEQ_LEN * FFN_DIM * 4);
  w.u    = (float*)alloc((size_t)SEQ_LEN * FFN_DIM * 4);
  w.sh    = (float*)alloc((size_t)N_SLOTS * D_MODEL * 4);
  w.hs    = (float*)alloc((size_t)N_SLOTS * D_MODEL * 4);
  w.newsh = (float*)alloc((size_t)N_SLOTS * D_MODEL * 4);
  w.gate  = (float*)alloc((size_t)N_SLOTS * D_MODEL * 4);
  w.tvec  = (float*)alloc((size_t)D_MODEL * 4);
  w.neg   = (float*)alloc((size_t)N_SLOTS * N_SLOTS * 4);
  w.div   = (float*)alloc(4);
  w.xn   = (unsigned short*)alloc((size_t)SEQ_LEN * D_MODEL * 2);
  w.kvn  = (unsigned short*)alloc((size_t)SEQ_LEN * D_MODEL * 2);
  w.o    = (unsigned short*)alloc((size_t)SEQ_LEN * D_MODEL * 2);
  w.hbf  = (unsigned short*)alloc((size_t)SEQ_LEN * FFN_DIM * 2);
  w.an   = (unsigned short*)alloc((size_t)N_SLOTS * D_MODEL * 2);
  w.pn   = (unsigned short*)alloc((size_t)N_SLOTS * D_MODEL * 2);

  // Decide if the swizzled-weight path fits in remaining scratch.
  const size_t wb_elems =
      (size_t)D_MODEL * D_MODEL +                 // qw
      2 * (size_t)D_MODEL * (N_GROUPS * HEAD_DIM) + // kw, vw
      (size_t)D_MODEL * D_MODEL +                 // ow
      2 * (size_t)D_MODEL * FFN_DIM +             // gw, uw
      (size_t)FFN_DIM * D_MODEL;                  // dw
  const size_t need_wb = (6 * wb_elems + (size_t)D_MODEL * D_MODEL) * 2 + 64 * 256;
  const bool big = (off + need_wb) <= ws_size;

  WBlk wblk[6];          // enc, dec, rsn0..3
  unsigned short* hgw_s = nullptr;
  const BlockP* bps[6] = { &enc, &dec, &rsn[0], &rsn[1], &rsn[2], &rsn[3] };
  if (big) {
    for (int b = 0; b < 6; ++b) {
      wblk[b].qw = (unsigned short*)alloc((size_t)D_MODEL * D_MODEL * 2);
      wblk[b].kw = (unsigned short*)alloc((size_t)D_MODEL * (N_GROUPS * HEAD_DIM) * 2);
      wblk[b].vw = (unsigned short*)alloc((size_t)D_MODEL * (N_GROUPS * HEAD_DIM) * 2);
      wblk[b].ow = (unsigned short*)alloc((size_t)D_MODEL * D_MODEL * 2);
      wblk[b].gw = (unsigned short*)alloc((size_t)D_MODEL * FFN_DIM * 2);
      wblk[b].uw = (unsigned short*)alloc((size_t)D_MODEL * FFN_DIM * 2);
      wblk[b].dw = (unsigned short*)alloc((size_t)FFN_DIM * D_MODEL * 2);
      swz(stream, bps[b]->qw, wblk[b].qw, D_MODEL, D_MODEL);
      swz(stream, bps[b]->kw, wblk[b].kw, D_MODEL, N_GROUPS * HEAD_DIM);
      swz(stream, bps[b]->vw, wblk[b].vw, D_MODEL, N_GROUPS * HEAD_DIM);
      swz(stream, bps[b]->ow, wblk[b].ow, D_MODEL, D_MODEL);
      swz(stream, bps[b]->gw, wblk[b].gw, D_MODEL, FFN_DIM);
      swz(stream, bps[b]->uw, wblk[b].uw, D_MODEL, FFN_DIM);
      swz(stream, bps[b]->dw, wblk[b].dw, FFN_DIM, D_MODEL);
    }
    hgw_s = (unsigned short*)alloc((size_t)D_MODEL * D_MODEL * 2);
    swz(stream, hgw, hgw_s, D_MODEL, D_MODEL);
  }
  const WBlk* wenc = big ? &wblk[0] : nullptr;
  const WBlk* wdec = big ? &wblk[1] : nullptr;
  const WBlk* wrsn[4] = { big ? &wblk[2] : nullptr, big ? &wblk[3] : nullptr,
                          big ? &wblk[4] : nullptr, big ? &wblk[5] : nullptr };

  // Setup.
  k_rope_tables<<<ROPE_ROWS, 32, 0, stream>>>(w.cosT, w.sinT);
  k_zero<<<1, 1, 0, stream>>>(w.div, 1);
  k_gather<<<SEQ_LEN, 256, 0, stream>>>(tokens, embed, w.x);
  k_copy<<<(N_SLOTS * D_MODEL + 255) / 256, 256, 0, stream>>>(w.sh, shared0, N_SLOTS * D_MODEL);

  // Shared-weight encoder (2 iterations, causal self-attention).
  for (int it = 0; it < 2; ++it)
    run_block(stream, w, enc, wenc, w.x, SEQ_LEN, nullptr, 0, true);

  // Reasoning steps.
  for (int step = 0; step < N_STEPS; ++step) {
    k_rmsnorm<<<1, 256, 0, stream>>>(time_emb + (size_t)step * D_MODEL, time_norm,
                                     w.tvec, D_MODEL, 1e-6f, 1);
    k_add_rowvec<<<(N_SLOTS * D_MODEL + 255) / 256, 256, 0, stream>>>(
        w.hs, w.sh, w.tvec, N_SLOTS * D_MODEL);
    for (int j = 0; j < 4; ++j) // slots cross-attend to sequence
      run_block(stream, w, rsn[j], wrsn[j], w.hs, N_SLOTS, w.x, SEQ_LEN, false);
    // Hunch gate: sigmoid(rms(h) @ hgw + hgb); blend shared/h.
    k_rmsnorm_bf<<<N_SLOTS, 256, 0, stream>>>(w.hs, hunch_norm, w.xn, D_MODEL, 1e-6f, 1);
    gemm_w(stream, w.xn, hgw, hgw_s, hgb, w.gate, N_SLOTS, D_MODEL, D_MODEL, 2);
    k_blend<<<(N_SLOTS * D_MODEL + 255) / 256, 256, 0, stream>>>(
        w.newsh, w.sh, w.hs, w.gate, N_SLOTS * D_MODEL);
    // InfoNCE(new_shared, shared): L2-normalize (bf16), neg = a @ a^T via WMMA.
    k_rmsnorm_bf<<<N_SLOTS, 256, 0, stream>>>(w.newsh, nullptr, w.an, D_MODEL, 1e-5f, 0);
    k_rmsnorm_bf<<<N_SLOTS, 256, 0, stream>>>(w.sh,    nullptr, w.pn, D_MODEL, 1e-5f, 0);
    { dim3 g(N_SLOTS / 16, N_SLOTS / 32);
      k_gemm_abt_bf<2><<<g, 32, 0, stream>>>(w.an, w.an, w.neg, N_SLOTS, N_SLOTS, D_MODEL); }
    k_infonce<<<1, 32, 0, stream>>>(w.an, w.pn, w.neg, raw_tau, w.div);
    k_copy<<<(N_SLOTS * D_MODEL + 255) / 256, 256, 0, stream>>>(
        w.sh, w.newsh, N_SLOTS * D_MODEL);
  }

  // Shared-weight decoder (2 iterations, cross-attend to reasoned slots).
  for (int it = 0; it < 2; ++it)
    run_block(stream, w, dec, wdec, w.x, SEQ_LEN, w.sh, N_SLOTS, false);

  // Tied-embedding head: logits = rms(x) @ embed^T (bandwidth-bound embed stream).
  float* out = (float*)d_out;
  k_rmsnorm_bf<<<SEQ_LEN, 256, 0, stream>>>(w.x, seq_norm, w.xn, D_MODEL, 1e-6f, 1);
  { dim3 g(SEQ_LEN / 16, VOCAB_SZ / 64);
    k_gemm_abt_f32<4><<<g, 32, 0, stream>>>(w.xn, embed, out, SEQ_LEN, VOCAB_SZ, D_MODEL); }

  // div_loss = mean(div / NSTEPS) over batch (B=1), appended after logits.
  k_finalize<<<1, 1, 0, stream>>>(w.div, out + (size_t)SEQ_LEN * VOCAB_SZ);
}
